// SparseMoE_45363444580493
// MI455X (gfx1250) — matrix operand order repaired and checked by reading, not compiled, not run on hardware
//
#include <hip/hip_runtime.h>
#include <cstdint>

#define DIM   1024
#define NTOK  8192
#define FFD   4096
#define NE    8
#define LDA   72           // padded LDS row stride (bf16 elems) for 64-wide K stage
#define KSTEP 64

typedef __bf16 bf16;
typedef bf16  v16bf __attribute__((ext_vector_type(16)));
typedef float v8f   __attribute__((ext_vector_type(8)));

#if __has_builtin(__builtin_amdgcn_global_load_async_to_lds_b128)
#define HAVE_ASYNC_LDS 1
typedef int v4i_ __attribute__((ext_vector_type(4)));
typedef __attribute__((address_space(1))) v4i_ g_v4i;
typedef __attribute__((address_space(3))) v4i_ l_v4i;
__device__ __forceinline__ void async_cp16(const bf16* g, bf16* l) {
  // 16B per lane, global -> LDS, tracked by ASYNCcnt
  __builtin_amdgcn_global_load_async_to_lds_b128((g_v4i*)g, (l_v4i*)l, 0, 0);
}
__device__ __forceinline__ void async_wait0() {
#if __has_builtin(__builtin_amdgcn_s_wait_asynccnt)
  __builtin_amdgcn_s_wait_asynccnt(0);
#else
  asm volatile("s_wait_asynccnt 0x0" ::: "memory");
#endif
}
#else
#define HAVE_ASYNC_LDS 0
#endif

// ---- fragment loaders (CDNA5 16-bit WMMA VGPR layouts, wave32) ----
// A 16x32 bf16: lane<16 holds K = {0..7, 16..23}; lane>=16 holds K = {8..15, 24..31}
__device__ __forceinline__ v16bf frag_a(const bf16* row, int kh) {
  v16bf r;
#pragma unroll
  for (int i = 0; i < 8; ++i) r[i] = row[kh * 8 + i];
#pragma unroll
  for (int i = 0; i < 8; ++i) r[8 + i] = row[16 + kh * 8 + i];
  return r;
}
// B 32x16 bf16: lane<16 (col n) holds K = {0..7, 16..23}; lane>=16 holds K = {8..15, 24..31}
__device__ __forceinline__ v16bf frag_b(const bf16* col, int kh) {
  v16bf r;
#pragma unroll
  for (int i = 0; i < 8; ++i) r[i] = col[kh * 8 + i];
#pragma unroll
  for (int i = 0; i < 8; ++i) r[8 + i] = col[16 + kh * 8 + i];
  return r;
}

__device__ __forceinline__ void cvt8(bf16* dd, float4 f0, float4 f1) {
  dd[0] = (bf16)f0.x; dd[1] = (bf16)f0.y; dd[2] = (bf16)f0.z; dd[3] = (bf16)f0.w;
  dd[4] = (bf16)f1.x; dd[5] = (bf16)f1.y; dd[6] = (bf16)f1.z; dd[7] = (bf16)f1.w;
}

struct BReg { float4 a, b, c, d; };   // 16 f32 staged in regs
__device__ __forceinline__ BReg load16f(const float* p) {
  BReg r;
  r.a = *(const float4*)(p);      r.b = *(const float4*)(p + 4);
  r.c = *(const float4*)(p + 8);  r.d = *(const float4*)(p + 12);
  return r;
}
__device__ __forceinline__ void store16f(bf16* dd, const BReg& r) {
  cvt8(dd, r.a, r.b); cvt8(dd + 8, r.c, r.d);
}

struct AReg { uint4 q[4]; };          // 32 bf16 staged in regs (fallback path)
__device__ __forceinline__ AReg load32h(const bf16* p) {
  AReg r;
#pragma unroll
  for (int i = 0; i < 4; ++i) r.q[i] = ((const uint4*)p)[i];
  return r;
}
__device__ __forceinline__ void store32h(bf16* d, const AReg& r) {
#pragma unroll
  for (int i = 0; i < 4; ++i) ((uint4*)d)[i] = r.q[i];
}

// -------------------- zero output + counters --------------------
__global__ void k_zero(float* __restrict__ out, int n,
                       int* __restrict__ counts, int* __restrict__ cursors) {
  int stride = gridDim.x * blockDim.x;
  for (int i = blockIdx.x * blockDim.x + threadIdx.x; i < n; i += stride) out[i] = 0.f;
  int i = blockIdx.x * blockDim.x + threadIdx.x;
  if (i < NE) { counts[i] = 0; cursors[i] = 0; }
}

// -------------------- router: gate + top2 + x->bf16 --------------------
__global__ void k_router(const float* __restrict__ x, const float* __restrict__ gw,
                         const float* __restrict__ gb, bf16* __restrict__ xb,
                         int* __restrict__ eIdx, float* __restrict__ eP,
                         int* __restrict__ counts) {
  int lane = threadIdx.x & 31;
  int wave = threadIdx.x >> 5;
  int token = blockIdx.x * 8 + wave;
  const float* xp = x + (size_t)token * DIM;
  float acc[NE];
#pragma unroll
  for (int e = 0; e < NE; ++e) acc[e] = 0.f;
  for (int d = lane; d < DIM; d += 32) {
    float xv = xp[d];
    xb[(size_t)token * DIM + d] = (bf16)xv;
#pragma unroll
    for (int e = 0; e < NE; ++e) acc[e] += xv * gw[e * DIM + d];
  }
#pragma unroll
  for (int e = 0; e < NE; ++e) {
#pragma unroll
    for (int off = 16; off > 0; off >>= 1)
      acc[e] += __shfl_xor(acc[e], off, 32);
  }
  if (lane == 0) {
    float lg[NE];
#pragma unroll
    for (int e = 0; e < NE; ++e) lg[e] = acc[e] + gb[e];
    int e0 = 0; float s0 = lg[0];
#pragma unroll
    for (int e = 1; e < NE; ++e) if (lg[e] > s0) { s0 = lg[e]; e0 = e; }
    int e1 = (e0 == 0) ? 1 : 0; float s1 = lg[e1];
#pragma unroll
    for (int e = 0; e < NE; ++e)
      if (e != e0 && lg[e] > s1) { s1 = lg[e]; e1 = e; }
    float p0 = 1.f / (1.f + __expf(s1 - s0));
    float p1 = 1.f - p0;
    eIdx[token * 2]     = e0;  eIdx[token * 2 + 1] = e1;
    eP[token * 2]       = p0;  eP[token * 2 + 1]   = p1;
    atomicAdd(&counts[e0], 1);
    atomicAdd(&counts[e1], 1);
  }
}

// -------------------- offsets + load-balancing loss --------------------
__global__ void k_scan(const int* __restrict__ counts, int* __restrict__ offsets,
                       float* __restrict__ out_loss) {
  if (threadIdx.x == 0) {
    int s = 0;
    for (int e = 0; e < NE; ++e) { offsets[e] = s; s += counts[e]; }
    offsets[NE] = s;
    float inv = 1.f / (float)s;
    float loss = 0.f;
    for (int e = 0; e < NE; ++e) {
      float f = (float)counts[e] * inv - 1.f / (float)NE;
      loss += f * f;
    }
    *out_loss = loss;
  }
}

// -------------------- pack (token, weight) rows per expert --------------------
__global__ void k_scatter(const int* __restrict__ eIdx, const float* __restrict__ eP,
                          const int* __restrict__ offsets, int* __restrict__ cursors,
                          int* __restrict__ rowToken, float* __restrict__ rowWeight) {
  int token = blockIdx.x * blockDim.x + threadIdx.x;
  if (token >= NTOK) return;
#pragma unroll
  for (int j = 0; j < 2; ++j) {
    int e = eIdx[token * 2 + j];
    int pos = atomicAdd(&cursors[e], 1);
    int r = offsets[e] + pos;
    rowToken[r]  = token;
    rowWeight[r] = eP[token * 2 + j];
  }
}

#define WMMA_BF16(A, B, C) \
  __builtin_amdgcn_wmma_f32_16x16x32_bf16(false, (A), false, (B), (short)0, (C), false, false)

// -------------------- GEMM1 compute stage (16 WMMAs) --------------------
__device__ __forceinline__ void ffn1_compute(const bf16* As, const bf16* Bs1,
                                             const bf16* Bs2, int wm, int wn,
                                             int row16, int kh,
                                             v8f (&c1)[2][2], v8f (&c2)[2][2]) {
#pragma unroll
  for (int kk = 0; kk < KSTEP; kk += 32) {
    const bf16* ap = &As[(wm * 32 + row16) * LDA + kk];
    v16bf a0 = frag_a(ap, kh);
    v16bf a1 = frag_a(ap + 16 * LDA, kh);
    const bf16* q1 = &Bs1[(wn * 32 + row16) * LDA + kk];
    v16bf p10 = frag_b(q1, kh);
    v16bf p11 = frag_b(q1 + 16 * LDA, kh);
    const bf16* q2 = &Bs2[(wn * 32 + row16) * LDA + kk];
    v16bf p20 = frag_b(q2, kh);
    v16bf p21 = frag_b(q2 + 16 * LDA, kh);

    c1[0][0] = WMMA_BF16(a0, p10, c1[0][0]);
    c1[0][1] = WMMA_BF16(a0, p11, c1[0][1]);
    c1[1][0] = WMMA_BF16(a1, p10, c1[1][0]);
    c1[1][1] = WMMA_BF16(a1, p11, c1[1][1]);
    c2[0][0] = WMMA_BF16(a0, p20, c2[0][0]);
    c2[0][1] = WMMA_BF16(a0, p21, c2[0][1]);
    c2[1][0] = WMMA_BF16(a1, p20, c2[1][0]);
    c2[1][1] = WMMA_BF16(a1, p21, c2[1][1]);
  }
}

// -------------------- GEMM1: H = silu(X w1^T + b1) * (X w2^T + b2) --------------------
// block tile 128(M) x 64(FF), K-stage 64, double-buffered software pipeline
__global__ __launch_bounds__(256) void k_ffn1(
    const bf16* __restrict__ xb, const float* __restrict__ w1,
    const float* __restrict__ b1, const float* __restrict__ w2,
    const float* __restrict__ b2, const int* __restrict__ counts,
    const int* __restrict__ offsets, const int* __restrict__ rowToken,
    bf16* __restrict__ H) {
  int e = blockIdx.z;
  int mcnt = counts[e];
  int mtile = blockIdx.y;
  if (mtile * 128 >= mcnt) return;
  int rowBase = offsets[e] + mtile * 128;
  int segEnd  = offsets[e] + mcnt;
  int ffbase  = blockIdx.x * 64;

  const int ASZ = 128 * LDA, BSZ = 64 * LDA;
  __shared__ __align__(16) bf16 As[2 * 128 * LDA];
  __shared__ __align__(16) bf16 Bs1[2 * 64 * LDA];
  __shared__ __align__(16) bf16 Bs2[2 * 64 * LDA];

  int t = threadIdx.x;
  int lane = t & 31, wid = t >> 5;
  int wm = wid >> 1, wn = wid & 1;          // wave tile: 32x32 within 128x64
  int row16 = lane & 15, kh = lane >> 4;

  // A staging: 128 rows x 64 cols bf16, 32 elems / thread
  int ar = t >> 1;
  int ac = (t & 1) * 32;
  int rA = rowBase + ar;
  int rAc = (rA < segEnd) ? rA : (segEnd - 1);
  const bf16* aSrc = xb + (size_t)rowToken[rAc] * DIM + ac;
  int aOff = ar * LDA + ac;
  // B staging: 64 rows(ff) x 64 cols f32 -> bf16, 16 elems / thread / matrix
  int br = t >> 2;
  int bc = (t & 3) * 16;
  const float* w1p = w1 + ((size_t)e * FFD + ffbase + br) * DIM + bc;
  const float* w2p = w2 + ((size_t)e * FFD + ffbase + br) * DIM + bc;
  int bOff = br * LDA + bc;

  v8f c1[2][2], c2[2][2];
#pragma unroll
  for (int i = 0; i < 2; ++i)
#pragma unroll
    for (int j = 0; j < 2; ++j)
#pragma unroll
      for (int v = 0; v < 8; ++v) { c1[i][j][v] = 0.f; c2[i][j][v] = 0.f; }

  // ---- prologue: stage 0 into buffer 0 ----
#if HAVE_ASYNC_LDS
#pragma unroll
  for (int q = 0; q < 4; ++q) async_cp16(aSrc + q * 8, &As[aOff + q * 8]);
#else
  { AReg ra = load32h(aSrc); store32h(&As[aOff], ra); }
#endif
  { BReg r1 = load16f(w1p), r2 = load16f(w2p);
    store16f(&Bs1[bOff], r1); store16f(&Bs2[bOff], r2); }
#if HAVE_ASYNC_LDS
  async_wait0();
#endif
  __syncthreads();

  for (int kb = 0; kb < DIM; kb += KSTEP) {
    int cur = (kb / KSTEP) & 1, nxt = cur ^ 1;
    bool more = (kb + KSTEP) < DIM;
#if HAVE_ASYNC_LDS
    if (more) {
#pragma unroll
      for (int q = 0; q < 4; ++q)
        async_cp16(aSrc + kb + KSTEP + q * 8, &As[nxt * ASZ + aOff + q * 8]);
    }
#else
    AReg ra;
    if (more) ra = load32h(aSrc + kb + KSTEP);
#endif
    BReg r1, r2;
    if (more) {
      r1 = load16f(w1p + kb + KSTEP);
      r2 = load16f(w2p + kb + KSTEP);
      __builtin_prefetch((const void*)(w1p + kb + 2 * KSTEP), 0, 1);
      __builtin_prefetch((const void*)(w2p + kb + 2 * KSTEP), 0, 1);
    }

    ffn1_compute(&As[cur * ASZ], &Bs1[cur * BSZ], &Bs2[cur * BSZ],
                 wm, wn, row16, kh, c1, c2);

    if (more) {
      store16f(&Bs1[nxt * BSZ + bOff], r1);
      store16f(&Bs2[nxt * BSZ + bOff], r2);
#if !HAVE_ASYNC_LDS
      store32h(&As[nxt * ASZ + aOff], ra);
#endif
    }
#if HAVE_ASYNC_LDS
    if (more) async_wait0();
#endif
    __syncthreads();
  }

#pragma unroll
  for (int i = 0; i < 2; ++i)
#pragma unroll
    for (int j = 0; j < 2; ++j) {
      int gn = ffbase + wn * 32 + j * 16 + row16;
      float bb1 = b1[e * FFD + gn];
      float bb2 = b2[e * FFD + gn];
#pragma unroll
      for (int v = 0; v < 8; ++v) {
        int mloc = wm * 32 + i * 16 + kh * 8 + v;
        int r = rowBase + mloc;
        if (r < segEnd) {
          float h1 = c1[i][j][v] + bb1;
          float h2 = c2[i][j][v] + bb2;
          float s  = h1 * __builtin_amdgcn_rcpf(1.f + __expf(-h1));  // silu
          H[(size_t)r * FFD + gn] = (bf16)(s * h2);
        }
      }
    }
}

// -------------------- GEMM2 compute stage (16 WMMAs) --------------------
__device__ __forceinline__ void ffn2_compute(const bf16* As, const bf16* Bs,
                                             int wm, int wn, int row16, int kh,
                                             v8f (&c)[2][4]) {
#pragma unroll
  for (int kk = 0; kk < KSTEP; kk += 32) {
    const bf16* ap = &As[(wm * 32 + row16) * LDA + kk];
    v16bf a0 = frag_a(ap, kh);
    v16bf a1 = frag_a(ap + 16 * LDA, kh);
    const bf16* q = &Bs[(wn * 64 + row16) * LDA + kk];
    v16bf p0 = frag_b(q, kh);
    v16bf p1 = frag_b(q + 16 * LDA, kh);
    v16bf p2 = frag_b(q + 32 * LDA, kh);
    v16bf p3 = frag_b(q + 48 * LDA, kh);

    c[0][0] = WMMA_BF16(a0, p0, c[0][0]);
    c[0][1] = WMMA_BF16(a0, p1, c[0][1]);
    c[0][2] = WMMA_BF16(a0, p2, c[0][2]);
    c[0][3] = WMMA_BF16(a0, p3, c[0][3]);
    c[1][0] = WMMA_BF16(a1, p0, c[1][0]);
    c[1][1] = WMMA_BF16(a1, p1, c[1][1]);
    c[1][2] = WMMA_BF16(a1, p2, c[1][2]);
    c[1][3] = WMMA_BF16(a1, p3, c[1][3]);
  }
}

// -------------------- GEMM2: out[token] += w * (H wp^T + bp) --------------------
// block tile 128(M) x 128(D), K-stage 64, double-buffered software pipeline
__global__ __launch_bounds__(256) void k_ffn2(
    const bf16* __restrict__ H, const float* __restrict__ wp,
    const float* __restrict__ bp, const int* __restrict__ counts,
    const int* __restrict__ offsets, const int* __restrict__ rowToken,
    const float* __restrict__ rowWeight, float* __restrict__ out) {
  int e = blockIdx.z;
  int mcnt = counts[e];
  int mtile = blockIdx.y;
  if (mtile * 128 >= mcnt) return;
  int rowBase = offsets[e] + mtile * 128;
  int segEnd  = offsets[e] + mcnt;
  int dbase   = blockIdx.x * 128;

  const int ASZ = 128 * LDA, BSZ = 128 * LDA;
  __shared__ __align__(16) bf16 As[2 * 128 * LDA];
  __shared__ __align__(16) bf16 Bs[2 * 128 * LDA];

  int t = threadIdx.x;
  int lane = t & 31, wid = t >> 5;
  int wm = wid >> 1, wn = wid & 1;          // wave tile: 32(M) x 64(N)
  int row16 = lane & 15, kh = lane >> 4;

  int ar = t >> 1;
  int ac = (t & 1) * 32;
  int rA = rowBase + ar;
  int rAc = (rA < segEnd) ? rA : (segEnd - 1);
  const bf16* aSrc = H + (size_t)rAc * FFD + ac;
  int aOff = ar * LDA + ac;
  // B staging: 128 rows(d) x 64 cols f32 -> bf16, 32 elems / thread
  int br = t >> 1;
  int bc = (t & 1) * 32;
  const float* wpp = wp + ((size_t)e * DIM + dbase + br) * FFD + bc;
  int bOff = br * LDA + bc;

  v8f c[2][4];
#pragma unroll
  for (int i = 0; i < 2; ++i)
#pragma unroll
    for (int j = 0; j < 4; ++j)
#pragma unroll
      for (int v = 0; v < 8; ++v) c[i][j][v] = 0.f;

  // ---- prologue ----
#if HAVE_ASYNC_LDS
#pragma unroll
  for (int q = 0; q < 4; ++q) async_cp16(aSrc + q * 8, &As[aOff + q * 8]);
#else
  { AReg ra = load32h(aSrc); store32h(&As[aOff], ra); }
#endif
  { BReg r0 = load16f(wpp), r1 = load16f(wpp + 16);
    store16f(&Bs[bOff], r0); store16f(&Bs[bOff + 16], r1); }
#if HAVE_ASYNC_LDS
  async_wait0();
#endif
  __syncthreads();

  for (int kb = 0; kb < FFD; kb += KSTEP) {
    int cur = (kb / KSTEP) & 1, nxt = cur ^ 1;
    bool more = (kb + KSTEP) < FFD;
#if HAVE_ASYNC_LDS
    if (more) {
#pragma unroll
      for (int q = 0; q < 4; ++q)
        async_cp16(aSrc + kb + KSTEP + q * 8, &As[nxt * ASZ + aOff + q * 8]);
    }
#else
    AReg ra;
    if (more) ra = load32h(aSrc + kb + KSTEP);
#endif
    BReg r0, r1;
    if (more) {
      r0 = load16f(wpp + kb + KSTEP);
      r1 = load16f(wpp + kb + KSTEP + 16);
      __builtin_prefetch((const void*)(wpp + kb + 2 * KSTEP), 0, 1);
    }

    ffn2_compute(&As[cur * ASZ], &Bs[cur * BSZ], wm, wn, row16, kh, c);

    if (more) {
      store16f(&Bs[nxt * BSZ + bOff], r0);
      store16f(&Bs[nxt * BSZ + bOff + 16], r1);
#if !HAVE_ASYNC_LDS
      store32h(&As[nxt * ASZ + aOff], ra);
#endif
    }
#if HAVE_ASYNC_LDS
    if (more) async_wait0();
#endif
    __syncthreads();
  }

#pragma unroll
  for (int i = 0; i < 2; ++i)
#pragma unroll
    for (int j = 0; j < 4; ++j) {
      int gn = dbase + wn * 64 + j * 16 + row16;
      float bbp = bp[e * DIM + gn];
#pragma unroll
      for (int v = 0; v < 8; ++v) {
        int mloc = wm * 32 + i * 16 + kh * 8 + v;
        int r = rowBase + mloc;
        if (r < segEnd) {
          int tok = rowToken[r];
          float wgt = rowWeight[r];
          atomicAdd(&out[(size_t)tok * DIM + gn], wgt * (c[i][j][v] + bbp));
        }
      }
    }
}

// -------------------- host launch --------------------
extern "C" void kernel_launch(void* const* d_in, const int* in_sizes, int n_in,
                              void* d_out, int out_size, void* d_ws, size_t ws_size,
                              hipStream_t stream) {
  (void)in_sizes; (void)n_in; (void)out_size; (void)ws_size;
  const float* x  = (const float*)d_in[0];
  const float* gw = (const float*)d_in[1];
  const float* gb = (const float*)d_in[2];
  const float* w1 = (const float*)d_in[3];
  const float* b1 = (const float*)d_in[4];
  const float* w2 = (const float*)d_in[5];
  const float* b2 = (const float*)d_in[6];
  const float* wp = (const float*)d_in[7];
  const float* bp = (const float*)d_in[8];
  float* out = (float*)d_out;

  char* ws = (char*)d_ws;
  int*   counts    = (int*)(ws + 0);
  int*   cursors   = (int*)(ws + 256);
  int*   offsets   = (int*)(ws + 512);
  int*   eIdx      = (int*)(ws + 1024);                       // 2N ints
  float* eP        = (float*)(ws + 1024 + 1 * 65536);         // 2N floats
  int*   rowToken  = (int*)(ws + 1024 + 2 * 65536);           // 2N ints
  float* rowWeight = (float*)(ws + 1024 + 3 * 65536);         // 2N floats
  bf16*  Xb        = (bf16*)(ws + 1024 + 4 * 65536);          // N*D bf16 (16 MB)
  bf16*  Hbuf      = (bf16*)(ws + 1024 + 4 * 65536 +
                             (size_t)NTOK * DIM * sizeof(bf16)); // 2N*FF bf16 (128 MB)

  k_zero<<<4096, 256, 0, stream>>>(out, NTOK * DIM + 1, counts, cursors);
  k_router<<<NTOK / 8, 256, 0, stream>>>(x, gw, gb, Xb, eIdx, eP, counts);
  k_scan<<<1, 32, 0, stream>>>(counts, offsets, out + (size_t)NTOK * DIM);
  k_scatter<<<NTOK / 256, 256, 0, stream>>>(eIdx, eP, offsets, cursors, rowToken, rowWeight);

  dim3 g1(FFD / 64, NTOK / 128, NE);    // (64, 64, 8), early-exit on empty tiles
  k_ffn1<<<g1, 256, 0, stream>>>(Xb, w1, b1, w2, b2, counts, offsets, rowToken, Hbuf);

  dim3 g2(DIM / 128, NTOK / 128, NE);   // (8, 64, 8)
  k_ffn2<<<g2, 256, 0, stream>>>(Hbuf, wp, bp, counts, offsets, rowToken, rowWeight, out);
}
